// MaskedWeight_22196390986035
// MI455X (gfx1250) — matrix operand order repaired and checked by reading, not compiled, not run on hardware
//
#include <hip/hip_runtime.h>
#include <cstdint>
#include <cstddef>

// ---- problem constants (match reference) ----
constexpr int IN_F  = 512;
constexpr int OUT_F = 512;
constexpr int BATCH = 64;
constexpr int DDIM  = 8;

typedef __attribute__((ext_vector_type(16))) __bf16 v16bf;
typedef __attribute__((ext_vector_type(8)))  float  v8f;
typedef __attribute__((ext_vector_type(4)))  unsigned int v4u;
typedef __attribute__((ext_vector_type(8)))  int  v8i;
typedef __attribute__((ext_vector_type(4)))  int  v4i;

// -----------------------------------------------------------------------------
// Kernel 1: per-output-row stats of the masked weight matrix.
// -----------------------------------------------------------------------------
__global__ void k_prep_rows(const float* __restrict__ W,
                            const float* __restrict__ diag_w,
                            float* __restrict__ rowscale,
                            float* __restrict__ gadd) {
  __shared__ float red[64];
  const int o  = blockIdx.x;      // 0..511
  const int t  = threadIdx.x;     // 0..63
  const int ob = o >> 6;
  float s = 0.f;
  for (int c = t; c < IN_F; c += 64) {
    const int cb = c >> 6;
    const float Wv = W[o * IN_F + c];
    float wv = 0.f;
    if (cb == ob)      wv = __expf(Wv);
    else if (cb < ob)  wv = Wv;
    s += wv * wv;
  }
  red[t] = s;
  __syncthreads();
  for (int off = 32; off > 0; off >>= 1) {
    if (t < off) red[t] += red[t + off];
    __syncthreads();
  }
  if (t == 0) {
    const float wsn = red[0];
    rowscale[o] = __expf(diag_w[o]) * rsqrtf(wsn);
    gadd[o]     = diag_w[o] - 0.5f * __logf(wsn);
  }
}

// -----------------------------------------------------------------------------
// Kernel 2: bf16 operands:  wT_bf[i*512+o] = bf16(rowscale[o]*w[o,i]),
//                           in_bf[b*512+i] = bf16(inputs[b,i])
// -----------------------------------------------------------------------------
__global__ void k_build_ops(const float* __restrict__ W,
                            const float* __restrict__ inputs,
                            const float* __restrict__ rowscale,
                            __bf16* __restrict__ wT_bf,
                            __bf16* __restrict__ in_bf) {
  const int idx = blockIdx.x * blockDim.x + threadIdx.x;
  const int N1 = IN_F * OUT_F;
  const int N2 = BATCH * IN_F;
  if (idx < N1) {
    const int i  = idx >> 9;
    const int o  = idx & 511;
    const int ob = o >> 6, ib = i >> 6;
    const float Wv = W[o * IN_F + i];
    const float wv = (ib == ob) ? __expf(Wv) : ((ib < ob) ? Wv : 0.f);
    wT_bf[idx] = (__bf16)(rowscale[o] * wv);
  } else if (idx < N1 + N2) {
    const int j = idx - N1;
    in_bf[j] = (__bf16)inputs[j];
  }
}

// -----------------------------------------------------------------------------
// Kernel 3: stabilized block-diagonal g:  M1, EG = exp(g - M1) (bf16)
// -----------------------------------------------------------------------------
__global__ void k_build_eg(const float* __restrict__ W,
                           const float* __restrict__ gadd,
                           float* __restrict__ M1,
                           __bf16* __restrict__ EG) {
  __shared__ float red[64];
  const int row = blockIdx.x;     // d*64+r
  const int c   = threadIdx.x;    // 0..63
  const int d   = row >> 6;
  const float gv = gadd[row] + W[row * IN_F + d * 64 + c];
  red[c] = gv;
  __syncthreads();
  for (int off = 32; off > 0; off >>= 1) {
    if (c < off) red[c] = fmaxf(red[c], red[c + off]);
    __syncthreads();
  }
  const float m = red[0];
  if (c == 0) M1[row] = m;
  EG[row * 64 + c] = (__bf16)__expf(gv - m);
}

// -----------------------------------------------------------------------------
// Kernel 4: out = inputs @ w.T + bias, LDS-staged bf16 WMMA GEMM.
// 16 blocks x 256 threads. Block bb owns a 64x32 output panel; 8 waves ->
// 4x2 grid of 16x16 tiles. K tiled in chunks of 128; B slice staged
// *transposed* (o-major, padded stride 144) so fragment gathers are
// contiguous 16B runs.
// -----------------------------------------------------------------------------
__global__ void k_gemm_out(const __bf16* __restrict__ in_bf,
                           const __bf16* __restrict__ wT_bf,
                           const float* __restrict__ bias,
                           float* __restrict__ out) {
  __shared__ __bf16 a_s[64 * 128];   // 16 KB  [row][k-chunk]
  __shared__ __bf16 b_s[32 * 144];   //  9 KB  [o_local][k-chunk], padded

  const int tid  = threadIdx.x;
  const int wave = tid >> 5;
  const int lane = tid & 31;
  const int n    = lane & 15;
  const int half = lane >> 4;
  const int bb   = blockIdx.x;              // output-column panel
  const int r0   = (wave & 3) << 4;
  const int ol0  = (wave >> 2) << 4;        // o offset within panel
  const int obase = bb * 32;

  v8f acc = {};
  for (int kc = 0; kc < IN_F; kc += 128) {
    __syncthreads();
    // stage A chunk (u32 = bf16 pairs, fully coalesced)
    const unsigned int* in_u = (const unsigned int*)in_bf;
    unsigned int* a_u = (unsigned int*)a_s;
    for (int idx = tid; idx < 4096; idx += 256) {
      const int row = idx >> 6, cp = idx & 63;
      a_u[row * 64 + cp] = in_u[row * 256 + (kc >> 1) + cp];
    }
    // stage B chunk transposed: b_s[o_local][kk]
    for (int idx = tid; idx < 4096; idx += 256) {
      const int kk = idx >> 5, ol = idx & 31;
      b_s[ol * 144 + kk] = wT_bf[(kc + kk) * OUT_F + obase + ol];
    }
    __syncthreads();
#pragma unroll
    for (int ks = 0; ks < 128; ks += 32) {
      v16bf a, b;
#pragma unroll
      for (int e = 0; e < 16; ++e) {
        const int kA = ks + e + half * 16;
        a[e] = a_s[(r0 + n) * 128 + kA];
        const int kB = ks + e + half * 16;
        b[e] = b_s[(ol0 + n) * 144 + kB];
      }
      acc = __builtin_amdgcn_wmma_f32_16x16x32_bf16(false, a, false, b,
                                                    (short)0, acc, false, false);
    }
  }
  const int o = obase + ol0 + n;
  const float bv = bias[o];
#pragma unroll
  for (int j = 0; j < 8; ++j)
    out[(r0 + j + half * 8) * OUT_F + o] = acc[j] + bv;
}

// -----------------------------------------------------------------------------
// Kernel 5: jac = M1 + M2 + log( EG @ exp(grad-M2) ), one (b,d) per block.
// grad (64x64 f32) and EG (64x64 bf16) tiles are fetched by the Tensor Data
// Mover (tensor_load_to_lds, D# per ISA 8.3/8.4), wave 0 waits TENSORcnt,
// then all 8 waves run the bf16 WMMA batched GEMM from LDS.
// -----------------------------------------------------------------------------
__global__ void k_gemm_jac(const float* __restrict__ grad,
                           const __bf16* __restrict__ EG,
                           const float* __restrict__ M1,
                           float* __restrict__ jac) {
  __shared__ float  lds_grad[64 * 64];   // 16 KB  grad[c][k]
  __shared__ __bf16 lds_eg  [64 * 64];   //  8 KB  EG[r][c]
  __shared__ __bf16 lds_egr [64 * 80];   // 10 KB  exp(grad-M2)[k][c] (padded)
  __shared__ float  lds_m2  [64];

  const int bd  = blockIdx.x;            // b*8 + d
  const int d   = bd & 7;
  const int tid = threadIdx.x;

  if (tid < 32) {
    // --- TDM: grad tile (64x64 f32, contiguous, stride 64) ---
    const unsigned long long ga =
        (unsigned long long)(uintptr_t)(grad + (size_t)bd * 4096);
    v4u g0;
    g0[0] = 1u;                                        // count=1, user D#
    g0[1] = (unsigned int)(uintptr_t)&lds_grad[0];     // lds_addr
    g0[2] = (unsigned int)ga;                          // global addr lo
    g0[3] = (unsigned int)((ga >> 32) & 0x1FFFFFFu) | (2u << 30); // hi|type=2
    v8i g1;
    g1[0] = (int)(2u << 16);       // data_size=2 -> 4-byte elements
    g1[1] = (int)(64u << 16);      // tensor_dim0 = 64 (lo16)
    g1[2] = (int)(64u << 16);      // dim0 hi=0 | tensor_dim1 lo16 = 64
    g1[3] = (int)(64u << 16);      // dim1 hi=0 | tile_dim0 = 64
    g1[4] = 64;                    // tile_dim1 = 64, tile_dim2 = 0
    g1[5] = 64;                    // tensor_dim0_stride = 64
    g1[6] = 0;
    g1[7] = 0;
    v4i z4 = {0, 0, 0, 0};
    v8i z8 = {0, 0, 0, 0, 0, 0, 0, 0};
    __builtin_amdgcn_tensor_load_to_lds(g0, g1, z4, z4, z8, 0);

    // --- TDM: EG tile (64x64 bf16, contiguous, stride 64) ---
    const unsigned long long ge =
        (unsigned long long)(uintptr_t)(EG + (size_t)d * 4096);
    g0[1] = (unsigned int)(uintptr_t)&lds_eg[0];
    g0[2] = (unsigned int)ge;
    g0[3] = (unsigned int)((ge >> 32) & 0x1FFFFFFu) | (2u << 30);
    g1[0] = (int)(1u << 16);       // data_size=1 -> 2-byte elements
    __builtin_amdgcn_tensor_load_to_lds(g0, g1, z4, z4, z8, 0);

    __builtin_amdgcn_s_wait_tensorcnt(0);
  }
  __syncthreads();

  if (tid < 64) {                              // M2[k] = max_c grad[c][k]
    float m = -3.402823466e38f;
    for (int c = 0; c < 64; ++c) m = fmaxf(m, lds_grad[c * 64 + tid]);
    lds_m2[tid] = m;
  }
  __syncthreads();

  // exp(grad - M2), stored transposed [k][c] with padded stride 80
  for (int idx = tid; idx < 4096; idx += 256) {
    const int c = idx >> 6, k = idx & 63;
    lds_egr[k * 80 + c] = (__bf16)__expf(lds_grad[c * 64 + k] - lds_m2[k]);
  }
  __syncthreads();

  const int wave = tid >> 5;
  const int lane = tid & 31;
  const int n    = lane & 15;
  const int half = lane >> 4;

#pragma unroll
  for (int it = 0; it < 2; ++it) {
    const int t  = wave * 2 + it;        // 16 tiles in a 4x4 grid
    const int r0 = (t >> 2) << 4;
    const int k0 = (t & 3) << 4;
    v8f acc = {};
#pragma unroll
    for (int ks = 0; ks < 64; ks += 32) {
      v16bf a, b;
#pragma unroll
      for (int e = 0; e < 16; ++e) {
        const int cA = ks + e + half * 16;
        a[e] = lds_eg[(r0 + n) * 64 + cA];
        const int cB = ks + e + half * 16;
        b[e] = lds_egr[(k0 + n) * 80 + cB];
      }
      acc = __builtin_amdgcn_wmma_f32_16x16x32_bf16(false, a, false, b,
                                                    (short)0, acc, false, false);
    }
    const float m2v = lds_m2[k0 + n];
#pragma unroll
    for (int j = 0; j < 8; ++j) {
      const int r = r0 + j + half * 8;
      jac[((size_t)bd * 64 + r) * 64 + (k0 + n)] =
          M1[d * 64 + r] + m2v + __logf(acc[j]);
    }
  }
}

// -----------------------------------------------------------------------------
// Launch wrapper
// -----------------------------------------------------------------------------
extern "C" void kernel_launch(void* const* d_in, const int* in_sizes, int n_in,
                              void* d_out, int out_size, void* d_ws, size_t ws_size,
                              hipStream_t stream) {
  const float* inputs = (const float*)d_in[0];   // (64,512)
  const float* grad   = (const float*)d_in[1];   // (64,8,64,64)
  const float* W      = (const float*)d_in[2];   // (512,512)
  const float* diag_w = (const float*)d_in[3];   // (512,1)
  const float* bias   = (const float*)d_in[4];   // (512,)

  char* ws = (char*)d_ws;
  float*  rowscale = (float*)(ws + 0);
  float*  gadd     = (float*)(ws + 2048);
  float*  M1       = (float*)(ws + 4096);
  __bf16* in_bf    = (__bf16*)(ws + 8192);                   // 64 KB
  __bf16* wT_bf    = (__bf16*)(ws + 8192 + 65536);           // 512 KB
  __bf16* EG       = (__bf16*)(ws + 8192 + 65536 + 524288);  // 64 KB

  float* out = (float*)d_out;             // 64*512
  float* jac = out + BATCH * OUT_F;       // 64*8*64*64

  k_prep_rows<<<OUT_F, 64, 0, stream>>>(W, diag_w, rowscale, gadd);

  const int total = IN_F * OUT_F + BATCH * IN_F;
  k_build_ops<<<(total + 255) / 256, 256, 0, stream>>>(W, inputs, rowscale,
                                                       wT_bf, in_bf);

  k_build_eg<<<OUT_F, 64, 0, stream>>>(W, gadd, M1, EG);

  k_gemm_out<<<16, 256, 0, stream>>>(in_bf, wT_bf, bias, out);

  k_gemm_jac<<<BATCH * DDIM, 256, 0, stream>>>(grad, EG, M1, jac);
}
